// MultiHeadAttention4UKT_21320217657945
// MI455X (gfx1250) — matrix operand order repaired and checked by reading, not compiled, not run on hardware
//
#include <hip/hip_runtime.h>
#include <hip/hip_bf16.h>
#include <math.h>

// Problem constants (from reference): B=16, S=512, D=512, H=8, DH=64
#define BB   16
#define SS   512
#define DD   512
#define HH   8
#define DHD  64
#define ROWS (BB * SS)   // 8192

typedef __bf16 bf16_t;
typedef __attribute__((ext_vector_type(16))) __bf16 v16bf;
typedef __attribute__((ext_vector_type(8)))  float  v8f;

static __device__ __forceinline__ unsigned short bf_bits(float f) {
  bf16_t h = (bf16_t)f;
  return __builtin_bit_cast(unsigned short, h);
}
static __device__ __forceinline__ unsigned int pack2(float a, float b) {
  return (unsigned int)bf_bits(a) | ((unsigned int)bf_bits(b) << 16);
}

// Load a 16-bf16 WMMA fragment from 32 CONTIGUOUS bytes (16B-aligned):
// fragment element e <-> K = e. Lowers to two ds_load_b128.
static __device__ __forceinline__ v16bf ld_frag32(const unsigned short* p) {
  union { v16bf v; struct { uint4 lo, hi; } q; } u;
  u.q.lo = *(const uint4*)(p);
  u.q.hi = *(const uint4*)(p + 8);
  return u.v;
}
// A-matrix fragment: two 8-element contiguous runs (K=8h.. and K=16+8h..).
static __device__ __forceinline__ v16bf ld_frag_split(const unsigned short* p0,
                                                      const unsigned short* p1) {
  union { v16bf v; struct { uint4 lo, hi; } q; } u;
  u.q.lo = *(const uint4*)(p0);
  u.q.hi = *(const uint4*)(p1);
  return u.v;
}

// ---------------------------------------------------------------------------
// Kernel W: convert the six 512x512 weight matrices to bf16 (same layout).
// Row-major W[j][c] is exactly the [n][k] layout the B-fragments want, so no
// transpose is needed anywhere: GEMM LDS staging is straight dword copies.
// ---------------------------------------------------------------------------
__global__ __launch_bounds__(256) void wconv_kernel(
    const float* __restrict__ w0, const float* __restrict__ w1,
    const float* __restrict__ w2, const float* __restrict__ w3,
    const float* __restrict__ w4, const float* __restrict__ w5,
    unsigned short* __restrict__ wt) {
  const float* src;
  switch (blockIdx.y) {
    case 0: src = w0; break; case 1: src = w1; break;
    case 2: src = w2; break; case 3: src = w3; break;
    case 4: src = w4; break; default: src = w5; break;
  }
  unsigned int* dst = (unsigned int*)(wt + (size_t)blockIdx.y * DD * DD);
  int p = blockIdx.x * blockDim.x + threadIdx.x;   // pair index
  float2 f = *(const float2*)(src + (size_t)2 * p);
  dst[p] = pack2(f.x, f.y);
}

// ---------------------------------------------------------------------------
// Kernel P: fused dual projection (mean & cov in one workgroup so the pack
// epilogue can build the Wasserstein row terms rq/rk = ||m||^2 + sum(c)).
// One workgroup = 64 rows x one head (64 cols), 8 waves:
//   wave w -> sel = w&1 (mean/cov GEMM), qsub = w>>1 (16-row subtile).
// LDS tiles: X [m][k] stride 40, W [n][k] stride 40; all fragment loads are
// 2x ds_load_b128; all staging is dword stores.
// isV==0: writes qcat/kcat [B,H,S,128] = [m | sqrt(clip(c,1e-24))] + row term
// isV==1: writes vm/vc TRANSPOSED [B,H,DH,S] bf16 (so AV staging is copies)
// ---------------------------------------------------------------------------
__global__ __launch_bounds__(256) void proj_kernel(
    const float* __restrict__ Xm, const float* __restrict__ Xc,
    const unsigned short* __restrict__ Wbm, const unsigned short* __restrict__ Wbc,
    const float* __restrict__ biasM, const float* __restrict__ biasC,
    unsigned short* __restrict__ outM, unsigned short* __restrict__ outC,
    float* __restrict__ rowTerm, int isV) {
  __shared__ unsigned short xm_t[64 * 40];   // [row][k], 80B rows (16-aligned)
  __shared__ unsigned short xc_t[64 * 40];
  __shared__ unsigned short wm_t[64 * 40];   // [n][k] == row-major W tile
  __shared__ unsigned short wc_t[64 * 40];
  __shared__ float rterm[64];

  const int tid  = threadIdx.x;
  const int lane = tid & 31, wv = tid >> 5;
  const int sel  = wv & 1,  qsub = wv >> 1;
  const int half = lane >> 4, ln = lane & 15;
  const int r0   = blockIdx.x * 64;
  const int h    = blockIdx.y;
  const int col0 = h * 64;

  v8f acc[4] = {};

  for (int ks = 0; ks < 16; ++ks) {
    const int k0 = ks * 32;
    __syncthreads();
    // X tiles: float2 -> packed bf16 dword (coalesced, dword LDS stores)
#pragma unroll
    for (int i = 0; i < 4; ++i) {
      int e = tid + 256 * i;                 // 1024 float2-pairs per tile
      int row = e >> 4, p = e & 15;
      float2 fm = *(const float2*)&Xm[(size_t)(r0 + row) * DD + k0 + 2 * p];
      float2 fc = *(const float2*)&Xc[(size_t)(r0 + row) * DD + k0 + 2 * p];
      ((unsigned int*)xm_t)[row * 20 + p] = pack2(fm.x, fm.y);
      ((unsigned int*)xc_t)[row * 20 + p] = pack2(fc.x, fc.y);
    }
    // W tiles: straight dword copies (W row-major == [n][k])
#pragma unroll
    for (int i = 0; i < 4; ++i) {
      int e = tid + 256 * i;                 // 1024 dwords per tile
      int j = e >> 4, c2 = e & 15;
      ((unsigned int*)wm_t)[j * 20 + c2] =
          ((const unsigned int*)(Wbm + (size_t)(col0 + j) * DD + k0))[c2];
      ((unsigned int*)wc_t)[j * 20 + c2] =
          ((const unsigned int*)(Wbc + (size_t)(col0 + j) * DD + k0))[c2];
    }
    __syncthreads();

    const unsigned short* xt = sel ? xc_t : xm_t;
    const unsigned short* wt = sel ? wc_t : wm_t;
    const unsigned short* xrow = xt + (qsub * 16 + ln) * 40;
    v16bf a = ld_frag_split(xrow + 8 * half, xrow + 16 + 8 * half);
#pragma unroll
    for (int ns = 0; ns < 4; ++ns) {
      v16bf bfr = ld_frag_split(wt + (ns * 16 + ln) * 40 + 8 * half,
                                wt + (ns * 16 + ln) * 40 + 16 + 8 * half);
      acc[ns] = __builtin_amdgcn_wmma_f32_16x16x32_bf16(
          false, a, false, bfr, (short)0, acc[ns], false, false);
    }
  }

  // bias epilogue
  const float* bias = sel ? biasC : biasM;
#pragma unroll
  for (int ns = 0; ns < 4; ++ns) {
    float bv = bias[col0 + ns * 16 + ln];
#pragma unroll
    for (int i = 0; i < 8; ++i) acc[ns][i] += bv;
  }

  if (isV) {
    // transposed value layout [B,H,DH,S]: hot AV staging becomes copies
    unsigned short* dst = sel ? outC : outM;
#pragma unroll
    for (int ns = 0; ns < 4; ++ns)
#pragma unroll
      for (int i = 0; i < 8; ++i) {
        int r = r0 + qsub * 16 + i + 8 * half;
        int bb = r >> 9, s = r & 511;
        int dh = ns * 16 + ln;
        size_t idx = (((size_t)(bb * HH + h)) * DHD + dh) * SS + s;
        dst[idx] = bf_bits(acc[ns][i]);
      }
  } else {
    // pack [m | sqrt(clip(c))] into qcat/kcat [B,H,S,128]
#pragma unroll
    for (int ns = 0; ns < 4; ++ns)
#pragma unroll
      for (int i = 0; i < 8; ++i) {
        int r = r0 + qsub * 16 + i + 8 * half;
        int bb = r >> 9, s = r & 511;
        size_t idx = (((size_t)(bb * HH + h)) * SS + s) * 128 + ns * 16 + ln;
        float v = acc[ns][i];
        if (sel) outM[idx + 64] = bf_bits(sqrtf(fmaxf(v, 1e-24f)));
        else     outM[idx]      = bf_bits(v);
      }
    // row term: sum over DH of mean^2 + cov (unclipped) -- ds_add_f32 reduce
    __syncthreads();
    if (tid < 64) rterm[tid] = 0.f;
    __syncthreads();
#pragma unroll
    for (int i = 0; i < 8; ++i) {
      float sum = 0.f;
#pragma unroll
      for (int ns = 0; ns < 4; ++ns) {
        float v = acc[ns][i];
        sum += sel ? v : v * v;
      }
      atomicAdd(&rterm[qsub * 16 + i + 8 * half], sum);
    }
    __syncthreads();
    if (tid < 64) {
      int r = r0 + tid;
      int bb = r >> 9, s = r & 511;
      rowTerm[((size_t)(bb * HH + h)) * SS + s] = rterm[tid];
    }
  }
}

// ---------------------------------------------------------------------------
// Kernel A: fused attention per (b,h,64-query block). Entire per-head K
// operand (512x136 bf16) + full 64x520 f32 score tile live in LDS (~294 KB
// of the 320 KB/WGP). Phases:
//  1) stage kcat/qcat/rk/rq into LDS (uint4 copies)
//  2) WMMA score GEMM (contraction 128): raw = (2G - rq - rk)/8
//  3) softmax#1 -> cumsum -> distance decay -> softmax#2
//     (max/sum/normalize passes parallel 4 threads/row; only cumsum serial)
//  4) WMMA AV: out_m = S @ vm, out_c = S^2 @ vc (V pre-transposed in global,
//     tiles staged with dword copies into the dead K region)
// ---------------------------------------------------------------------------
__global__ __launch_bounds__(256) void attn_kernel(
    const unsigned short* __restrict__ qcat, const unsigned short* __restrict__ kcat,
    const float* __restrict__ rq, const float* __restrict__ rk,
    const unsigned short* __restrict__ vmp, const unsigned short* __restrict__ vcp,
    const float* __restrict__ gammas, const int* __restrict__ zero_pad,
    unsigned short* __restrict__ catm, unsigned short* __restrict__ catc) {
  extern __shared__ char smem[];
  unsigned short* kc_l = (unsigned short*)smem;   // [512][136] (272B rows)
  unsigned short* qc_l = kc_l + 512 * 136;        // [64][136]
  float* sc   = (float*)(qc_l + 64 * 136);        // [64][520]
  float* rk_l = sc + 64 * 520;                    // [512]
  float* rq_l = rk_l + 512;                       // [64]
  float* pmax = rq_l + 64;                        // [64][4]
  float* psum = pmax + 256;                       // [64][4]
  float* m2s  = psum + 256;                       // [64]
  unsigned short* vm_l = kc_l;                    // reuse: [64 dh][40] tiles
  unsigned short* vc_l = kc_l + 64 * 40;

  const int tid  = threadIdx.x;
  const int lane = tid & 31, wv = tid >> 5;
  const int half = lane >> 4, ln = lane & 15;
  const int qb = blockIdx.x * 64;
  const int h  = blockIdx.y, bb = blockIdx.z;
  const size_t headBase = ((size_t)(bb * HH + h)) * SS;

  // ---- phase 1: stage operands (uint4 copies) ----
  const uint4* kglob = (const uint4*)(kcat + headBase * 128);
  __builtin_prefetch(kglob, 0, 1);                // global_prefetch_b8
#pragma unroll 4
  for (int i = 0; i < 32; ++i) {
    int e = tid + 256 * i;                        // 8192 uint4
    int j = e >> 4, c4 = e & 15;
    *(uint4*)((unsigned int*)kc_l + j * 68 + c4 * 4) = kglob[e];
  }
  const uint4* qglob = (const uint4*)(qcat + (headBase + qb) * 128);
#pragma unroll
  for (int i = 0; i < 4; ++i) {
    int e = tid + 256 * i;                        // 1024 uint4
    int j = e >> 4, c4 = e & 15;
    *(uint4*)((unsigned int*)qc_l + j * 68 + c4 * 4) = qglob[e];
  }
#pragma unroll
  for (int i = 0; i < 2; ++i) {
    int e = tid + 256 * i;
    rk_l[e] = rk[headBase + e];
  }
  if (tid < 64) rq_l[tid] = rq[headBase + qb + tid];
  __syncthreads();

  // ---- phase 2: score GEMM, 64x512, contraction 128 ----
  {
    const int qsub = wv >> 1, kpar = wv & 1;
    const unsigned short* qrow = qc_l + (qsub * 16 + ln) * 136;
    v16bf a[4];
#pragma unroll
    for (int cs = 0; cs < 4; ++cs)
      a[cs] = ld_frag_split(qrow + cs * 32 + 8 * half,
                            qrow + cs * 32 + 16 + 8 * half);
    for (int t = 0; t < 16; ++t) {
      const int j0 = (kpar + 2 * t) * 16;
      const unsigned short* krow = kc_l + (j0 + ln) * 136 + 8 * half;
      v8f acc = {};
#pragma unroll
      for (int cs = 0; cs < 4; ++cs) {
        v16bf bfr = ld_frag_split(krow + cs * 32, krow + cs * 32 + 16);
        acc = __builtin_amdgcn_wmma_f32_16x16x32_bf16(
            false, a[cs], false, bfr, (short)0, acc, false, false);
      }
      float rkv = rk_l[j0 + ln];
#pragma unroll
      for (int i = 0; i < 8; ++i) {
        int ml = qsub * 16 + i + 8 * half;
        sc[ml * 520 + j0 + ln] = (2.f * acc[i] - rq_l[ml] - rkv) * 0.125f;
      }
    }
  }
  __syncthreads();

  // ---- phase 3: softmax / cumsum / decay / softmax (causal) ----
  const int row = tid >> 2, sub = tid & 3;
  const int qgr = qb + row, Lr = qgr + 1;
  const int kb = sub * 128;
  const int ke = (kb + 128 < Lr) ? (kb + 128) : Lr;
  {
    float* sr = sc + row * 520;
    float pm = -3.0e38f;
    for (int k = kb; k < ke; ++k) pm = fmaxf(pm, sr[k]);
    pmax[row * 4 + sub] = pm;
  }
  __syncthreads();
  {
    const float* pr = pmax + row * 4;
    float m1 = fmaxf(fmaxf(pr[0], pr[1]), fmaxf(pr[2], pr[3]));
    float* sr = sc + row * 520;
    float ps = 0.f;
    for (int k = kb; k < ke; ++k) ps += expf(sr[k] - m1);
    psum[row * 4 + sub] = ps;
  }
  __syncthreads();
  if (tid < 64) {                       // serial cumsum + decay (per row)
    float* sr = sc + tid * 520;
    const int qg = qb + tid, L = qg + 1;
    const float* pr = pmax + tid * 4;
    float m1 = fmaxf(fmaxf(pr[0], pr[1]), fmaxf(pr[2], pr[3]));
    const float* ps = psum + tid * 4;
    float inv1 = 1.f / (ps[0] + ps[1] + ps[2] + ps[3]);
    float g  = gammas[h];
    float gam = -((g > 20.f) ? g : log1pf(expf(g)));   // -softplus
    float cum = 0.f, m2 = -3.0e38f;
    for (int k = 0; k < L; ++k) {
      float s = expf(sr[k] - m1) * inv1;
      cum += s;
      float rem = 1.f - cum; if (rem < 0.f) rem = 0.f;  // disttot == 1
      float dist = sqrtf(rem * (float)(qg - k));        // pos = |q-k|
      float te = expf(dist * gam);
      te = fminf(fmaxf(te, 1e-5f), 1e5f);
      float r2 = sr[k] * te;
      sr[k] = r2;
      m2 = fmaxf(m2, r2);
    }
    m2s[tid] = m2;
  }
  __syncthreads();
  {
    float m2 = m2s[row];
    float* sr = sc + row * 520;
    float ps = 0.f;
    for (int k = kb; k < ke; ++k) ps += expf(sr[k] - m2);
    psum[row * 4 + sub] = ps;
  }
  __syncthreads();
  {
    float m2 = m2s[row];
    const float* ps = psum + row * 4;
    float scale = 1.f / (ps[0] + ps[1] + ps[2] + ps[3]);
    if (zero_pad[0] != 0 && qgr == 0) scale = 0.f;      // zero first query row
    float* sr = sc + row * 520;
    for (int k = kb; k < kb + 128; ++k)
      sr[k] = (k < Lr) ? expf(sr[k] - m2) * scale : 0.f;
  }
  __syncthreads();

  // ---- phase 4: AV GEMMs (contraction 512); squared scores for cov path ----
  {
    const int mat = wv & 1, qsub = wv >> 1;             // 0: mean, 1: cov
    const size_t vBase = ((size_t)(bb * HH + h)) * DHD; // [B,H,DH,S] layout
    v8f acc[4] = {};
    for (int ks = 0; ks < 16; ++ks) {
      const int k0 = ks * 32;
      __syncthreads();
      // V tiles: straight dword copies (global already [dh][key])
#pragma unroll
      for (int i = 0; i < 4; ++i) {
        int e = tid + 256 * i;                          // 1024 dwords each
        int dh = e >> 4, c2 = e & 15;
        ((unsigned int*)vm_l)[dh * 20 + c2] =
            ((const unsigned int*)(vmp + (vBase + dh) * SS + k0))[c2];
        ((unsigned int*)vc_l)[dh * 20 + c2] =
            ((const unsigned int*)(vcp + (vBase + dh) * SS + k0))[c2];
      }
      __syncthreads();
      // A fragment from f32 scores (float4 loads, square for cov in f32)
      const float* sp = sc + (qsub * 16 + ln) * 520 + k0;
      float4 f0 = *(const float4*)(sp + 8 * half);
      float4 f1 = *(const float4*)(sp + 8 * half + 4);
      float4 f2 = *(const float4*)(sp + 16 + 8 * half);
      float4 f3 = *(const float4*)(sp + 16 + 8 * half + 4);
      float fv[16] = {f0.x, f0.y, f0.z, f0.w, f1.x, f1.y, f1.z, f1.w,
                      f2.x, f2.y, f2.z, f2.w, f3.x, f3.y, f3.z, f3.w};
      v16bf a;
#pragma unroll
      for (int e = 0; e < 16; ++e) {
        float x = fv[e];
        if (mat) x *= x;
        a[e] = (bf16_t)x;
      }
      const unsigned short* vt = mat ? vc_l : vm_l;
#pragma unroll
      for (int ns = 0; ns < 4; ++ns) {
        v16bf bfr = ld_frag_split(vt + (ns * 16 + ln) * 40 + 8 * half,
                                  vt + (ns * 16 + ln) * 40 + 16 + 8 * half);
        acc[ns] = __builtin_amdgcn_wmma_f32_16x16x32_bf16(
            false, a, false, bfr, (short)0, acc[ns], false, false);
      }
    }
    unsigned short* dst = mat ? catc : catm;
#pragma unroll
    for (int ns = 0; ns < 4; ++ns)
#pragma unroll
      for (int i = 0; i < 8; ++i) {
        int s = qb + qsub * 16 + i + 8 * half;
        size_t idx = ((size_t)bb * SS + s) * DD + h * DHD + ns * 16 + ln;
        dst[idx] = bf_bits(acc[ns][i]);
      }
  }
}

// ---------------------------------------------------------------------------
// Kernel O: output projection  out = cat @ Wo^T + bo  (f32 result to d_out).
// Workgroup: 64 rows x 64 cols, 8 waves (4 row subtiles x 2 col halves).
// ---------------------------------------------------------------------------
__global__ __launch_bounds__(256) void oproj_kernel(
    const unsigned short* __restrict__ X, const unsigned short* __restrict__ Wb,
    const float* __restrict__ bias, float* __restrict__ out) {
  __shared__ unsigned short x_t[64 * 40];   // [row][k]
  __shared__ unsigned short w_t[64 * 40];   // [n][k] == row-major W tile
  const int tid = threadIdx.x, lane = tid & 31, wv = tid >> 5;
  const int half = lane >> 4, ln = lane & 15;
  const int qsub = wv & 3, nh = wv >> 2;
  const int r0 = blockIdx.x * 64, col0 = blockIdx.y * 64;
  v8f acc[2] = {};
  for (int ks = 0; ks < 16; ++ks) {
    const int k0 = ks * 32;
    __syncthreads();
    {
      // X tile: one uint4 (8 bf16) per thread, 256 uint4 total
      int rrow = tid >> 2, c4 = tid & 3;
      *(uint4*)((unsigned int*)x_t + rrow * 20 + c4 * 4) =
          *(const uint4*)(X + (size_t)(r0 + rrow) * DD + k0 + c4 * 8);
    }
    {
      // W tile: one uint4 per thread (row-major copy)
      int j = tid >> 2, c4 = tid & 3;
      *(uint4*)((unsigned int*)w_t + j * 20 + c4 * 4) =
          *(const uint4*)(Wb + (size_t)(col0 + j) * DD + k0 + c4 * 8);
    }
    __syncthreads();
    const unsigned short* xrow = x_t + (qsub * 16 + ln) * 40;
    v16bf a = ld_frag_split(xrow + 8 * half, xrow + 16 + 8 * half);
#pragma unroll
    for (int t = 0; t < 2; ++t) {
      int ns = nh * 2 + t;
      v16bf bfr = ld_frag_split(w_t + (ns * 16 + ln) * 40 + 8 * half,
                                w_t + (ns * 16 + ln) * 40 + 16 + 8 * half);
      acc[t] = __builtin_amdgcn_wmma_f32_16x16x32_bf16(
          false, a, false, bfr, (short)0, acc[t], false, false);
    }
  }
#pragma unroll
  for (int t = 0; t < 2; ++t) {
    int ns = nh * 2 + t;
    float bv = bias[col0 + ns * 16 + ln];
#pragma unroll
    for (int i = 0; i < 8; ++i) {
      int r = r0 + qsub * 16 + i + 8 * half;
      out[(size_t)r * DD + col0 + ns * 16 + ln] = acc[t][i] + bv;
    }
  }
}

// ---------------------------------------------------------------------------
extern "C" void kernel_launch(void* const* d_in, const int* in_sizes, int n_in,
                              void* d_out, int out_size, void* d_ws, size_t ws_size,
                              hipStream_t stream) {
  (void)in_sizes; (void)n_in; (void)out_size; (void)ws_size;
  const float* q_mean  = (const float*)d_in[0];
  const float* q_cov   = (const float*)d_in[1];
  const float* k_mean  = (const float*)d_in[2];
  const float* k_cov   = (const float*)d_in[3];
  const float* v_mean  = (const float*)d_in[4];
  const float* v_cov   = (const float*)d_in[5];
  const float* Wk_mean = (const float*)d_in[6];
  const float* bk_mean = (const float*)d_in[7];
  const float* Wk_cov  = (const float*)d_in[8];
  const float* bk_cov  = (const float*)d_in[9];
  const float* Wv_mean = (const float*)d_in[10];
  const float* bv_mean = (const float*)d_in[11];
  const float* Wv_cov  = (const float*)d_in[12];
  const float* bv_cov  = (const float*)d_in[13];
  const float* Wo_mean = (const float*)d_in[14];
  const float* bo_mean = (const float*)d_in[15];
  const float* Wo_cov  = (const float*)d_in[16];
  const float* bo_cov  = (const float*)d_in[17];
  const float* gammas  = (const float*)d_in[18];
  const int*   zero_pad = (const int*)d_in[20];
  float* out = (float*)d_out;

  // workspace partition (256-byte aligned)
  unsigned char* ws = (unsigned char*)d_ws;
  size_t off = 0;
  auto take = [&](size_t bytes) {
    void* p = ws + off;
    off = (off + bytes + 255) & ~(size_t)255;
    return p;
  };
  unsigned short* wt   = (unsigned short*)take((size_t)6 * DD * DD * 2);   // 3 MB
  unsigned short* qcat = (unsigned short*)take((size_t)BB * HH * SS * 128 * 2);
  unsigned short* kcat = (unsigned short*)take((size_t)BB * HH * SS * 128 * 2);
  float* rq = (float*)take((size_t)BB * HH * SS * 4);
  float* rk = (float*)take((size_t)BB * HH * SS * 4);
  unsigned short* vm   = (unsigned short*)take((size_t)BB * HH * SS * DHD * 2);
  unsigned short* vc   = (unsigned short*)take((size_t)BB * HH * SS * DHD * 2);
  unsigned short* catm = (unsigned short*)take((size_t)BB * SS * DD * 2);
  unsigned short* catc = (unsigned short*)take((size_t)BB * SS * DD * 2);

  unsigned short* wt_km = wt + 0 * (size_t)DD * DD;  // bf16(Wk_mean)
  unsigned short* wt_kc = wt + 1 * (size_t)DD * DD;  // bf16(Wk_cov)
  unsigned short* wt_vm = wt + 2 * (size_t)DD * DD;  // bf16(Wv_mean)
  unsigned short* wt_vc = wt + 3 * (size_t)DD * DD;  // bf16(Wv_cov)
  unsigned short* wt_om = wt + 4 * (size_t)DD * DD;  // bf16(Wo_mean)
  unsigned short* wt_oc = wt + 5 * (size_t)DD * DD;  // bf16(Wo_cov)

  // W: bf16 convert (no transpose needed; row-major W == B-fragment layout)
  wconv_kernel<<<dim3(DD * DD / 2 / 256, 6), 256, 0, stream>>>(
      Wk_mean, Wk_cov, Wv_mean, Wv_cov, Wo_mean, Wo_cov, wt);

  // P: projections (Q and K share the key weights, per key_query_same=True)
  dim3 pg(ROWS / 64, HH);
  proj_kernel<<<pg, 256, 0, stream>>>(q_mean, q_cov, wt_km, wt_kc,
                                      bk_mean, bk_cov, qcat, nullptr, rq, 0);
  proj_kernel<<<pg, 256, 0, stream>>>(k_mean, k_cov, wt_km, wt_kc,
                                      bk_mean, bk_cov, kcat, nullptr, rk, 0);
  proj_kernel<<<pg, 256, 0, stream>>>(v_mean, v_cov, wt_vm, wt_vc,
                                      bv_mean, bv_cov, vm, vc, nullptr, 1);

  // A: fused attention. Dynamic LDS: K(512x136 bf16) + Q(64x136 bf16)
  //    + scores(64x520 f32) + reduction scratch = 294,400 B (< 320 KB/WGP)
  const size_t smemA = (size_t)(512 * 136 + 64 * 136) * 2 +
                       (size_t)(64 * 520 + 512 + 64 + 256 + 256 + 64) * 4;
  attn_kernel<<<dim3(SS / 64, HH, BB), 256, smemA, stream>>>(
      qcat, kcat, rq, rk, vm, vc, gammas, zero_pad, catm, catc);

  // O: output projections (mean first, then cov — tuple return order)
  dim3 og(ROWS / 64, DD / 64);
  oproj_kernel<<<og, 256, 0, stream>>>(catm, wt_om, bo_mean, out);
  oproj_kernel<<<og, 256, 0, stream>>>(catc, wt_oc, bo_cov, out + (size_t)ROWS * DD);
}
